// Model_41343355191640
// MI455X (gfx1250) — matrix operand order repaired and checked by reading, not compiled, not run on hardware
//
#include <hip/hip_runtime.h>
#include <hip/hip_bf16.h>

// CDNA5 / gfx1250 wave32 WMMA implementation of the CovPool steganalysis net.
// - conv K-order permuted to (tap, ci): one 3x3 tap per 32-wide K step,
//   constant-stride branchless gathers (clamped address + cndmask select).
// - packed conv weights staged into LDS via the Tensor Data Mover
//   (TENSOR_LOAD_TO_LDS + s_wait_tensorcnt), manual-copy fallback.
// - Newton-Schulz runs on V_WMMA_F32_16X16X4_F32 (pure fp32).

typedef __attribute__((ext_vector_type(16))) _Float16 v16h;
typedef __attribute__((ext_vector_type(8)))  float    v8f;

#if __has_builtin(__builtin_amdgcn_wmma_f32_16x16x4_f32)
#define HAVE_WMMA_F32X4 1
typedef __attribute__((ext_vector_type(2))) float v2f;
#endif

#if __has_builtin(__builtin_amdgcn_tensor_load_to_lds) && \
    __has_builtin(__builtin_amdgcn_s_wait_tensorcnt)
#define HAVE_TDM 1
typedef __attribute__((ext_vector_type(4))) unsigned int u32x4;
typedef __attribute__((ext_vector_type(4))) int          i32x4;
typedef __attribute__((ext_vector_type(8))) int          i32x8;
#endif

#define BN_EPS 1e-5f

__device__ __forceinline__ v16h load_a_frag_f32(const float* p, int hi) {
  // A-frag local K: [hi*8, hi*8+8) and [16+hi*8, 16+hi*8+8) - two contiguous spans.
  const float4* q = (const float4*)(p + hi * 8);
  float4 a0 = q[0], a1 = q[1];
  const float4* r = (const float4*)(p + 16 + hi * 8);
  float4 a2 = r[0], a3 = r[1];
  v16h f;
  f[0] = (_Float16)a0.x;  f[1] = (_Float16)a0.y;  f[2] = (_Float16)a0.z;  f[3] = (_Float16)a0.w;
  f[4] = (_Float16)a1.x;  f[5] = (_Float16)a1.y;  f[6] = (_Float16)a1.z;  f[7] = (_Float16)a1.w;
  f[8] = (_Float16)a2.x;  f[9] = (_Float16)a2.y;  f[10] = (_Float16)a2.z; f[11] = (_Float16)a2.w;
  f[12] = (_Float16)a3.x; f[13] = (_Float16)a3.y; f[14] = (_Float16)a3.z; f[15] = (_Float16)a3.w;
  return f;
}

__device__ __forceinline__ v16h load_b_frag_f32(const float* p, int hi) {
  // B-frag local K: [hi*8, hi*8+8) and [16+hi*8, 16+hi*8+8) - two contiguous spans.
  const float4* q = (const float4*)(p + hi * 8);
  float4 b0 = q[0], b1 = q[1];
  const float4* r = (const float4*)(p + 16 + hi * 8);
  float4 b2 = r[0], b3 = r[1];
  v16h f;
  f[0] = (_Float16)b0.x;  f[1] = (_Float16)b0.y;  f[2] = (_Float16)b0.z;  f[3] = (_Float16)b0.w;
  f[4] = (_Float16)b1.x;  f[5] = (_Float16)b1.y;  f[6] = (_Float16)b1.z;  f[7] = (_Float16)b1.w;
  f[8] = (_Float16)b2.x;  f[9] = (_Float16)b2.y;  f[10] = (_Float16)b2.z; f[11] = (_Float16)b2.w;
  f[12] = (_Float16)b3.x; f[13] = (_Float16)b3.y; f[14] = (_Float16)b3.z; f[15] = (_Float16)b3.w;
  return f;
}

// ---------------------------------------------------------------------------
// Conv-weight repack: f32 [Cout x (Cin*9)] (k_src = ci*9 + tap) -> WMMA A tiles
// with permuted K: k = tap*Cinp + ci (Cinp = Cin rounded to 32), f16.
// ---------------------------------------------------------------------------
__global__ void repack_conv_kernel(const float* __restrict__ W, _Float16* __restrict__ P,
                                   int M, int Cin, int Cinp) {
  int Kpad = 9 * Cinp;
  int idx = blockIdx.x * blockDim.x + threadIdx.x;
  if (idx >= M * Kpad) return;
  int k = idx % Kpad;
  int m = idx / Kpad;
  int t = k / Cinp;
  int ci = k - t * Cinp;
  float v = (ci < Cin) ? W[(size_t)m * (Cin * 9) + ci * 9 + t] : 0.0f;
  int tm = m >> 4, mlo = m & 15;
  int tk = k >> 5, kl = k & 31;
  int hi = kl >> 4;
  int e = kl & 15;
  int tkcnt = Kpad >> 5;
  P[(((size_t)(tm * tkcnt + tk)) * 32 + (mlo + (hi << 4))) * 16 + e] = (_Float16)v;
}

// Flat repack (fc10): f32 [M x K] row-major, K order unchanged.
__global__ void repack_flat_kernel(const float* __restrict__ W, _Float16* __restrict__ P,
                                   int M, int K) {
  int idx = blockIdx.x * blockDim.x + threadIdx.x;
  if (idx >= M * K) return;
  int k = idx % K;
  int m = idx / K;
  float v = W[(size_t)m * K + k];
  int tm = m >> 4, mlo = m & 15;
  int tk = k >> 5, kl = k & 31;
  int g = kl >> 3, hi = g & 1;
  int e = (kl & 7) + ((g >> 1) << 3);
  int tkcnt = K >> 5;
  P[(((size_t)(tm * tkcnt + tk)) * 32 + (mlo + (hi << 4))) * 16 + e] = (_Float16)v;
}

// ---------------------------------------------------------------------------
// HPF: 5x5 conv (30 filters -> 32-channel zero-padded output) + clamp [-3,3].
// ---------------------------------------------------------------------------
__global__ void hpf_kernel(const float* __restrict__ x, const float* __restrict__ w,
                           float* __restrict__ out, int B, int H, int W) {
  size_t idx = (size_t)blockIdx.x * blockDim.x + threadIdx.x;
  size_t npix = (size_t)H * W;
  if (idx >= (size_t)B * 32 * npix) return;
  int p = (int)(idx % npix);
  int f = (int)((idx / npix) % 32);
  int b = (int)(idx / (npix * 32));
  if (f >= 30) { out[idx] = 0.0f; return; }
  int oh = p / W, ow = p % W;
  float s = 0.0f;
  #pragma unroll
  for (int kh = 0; kh < 5; ++kh) {
    int ih = oh + kh - 2;
    if ((unsigned)ih >= (unsigned)H) continue;
    #pragma unroll
    for (int kw = 0; kw < 5; ++kw) {
      int iw = ow + kw - 2;
      if ((unsigned)iw >= (unsigned)W) continue;
      s += w[f * 25 + kh * 5 + kw] * x[((size_t)b * H + ih) * W + iw];
    }
  }
  s = s > 3.0f ? 3.0f : (s < -3.0f ? -3.0f : s);
  out[idx] = s;
}

// AvgPool2d(3, stride 2, pad 1), count_include_pad (always /9).
__global__ void avgpool_kernel(const float* __restrict__ in, float* __restrict__ out,
                               int B, int C, int Hin, int Win) {
  int Ho = Hin >> 1, Wo = Win >> 1;
  size_t idx = (size_t)blockIdx.x * blockDim.x + threadIdx.x;
  if (idx >= (size_t)B * C * Ho * Wo) return;
  int ow = (int)(idx % Wo);
  int oh = (int)((idx / Wo) % Ho);
  int c  = (int)((idx / ((size_t)Wo * Ho)) % C);
  int b  = (int)(idx / ((size_t)Wo * Ho * C));
  const float* src = in + ((size_t)b * C + c) * Hin * Win;
  float s = 0.0f;
  #pragma unroll
  for (int i = 0; i < 3; ++i) {
    int ih = 2 * oh - 1 + i;
    if ((unsigned)ih >= (unsigned)Hin) continue;
    #pragma unroll
    for (int j = 0; j < 3; ++j) {
      int iw = 2 * ow - 1 + j;
      if ((unsigned)iw >= (unsigned)Win) continue;
      s += src[(size_t)ih * Win + iw];
    }
  }
  out[idx] = s * (1.0f / 9.0f);
}

// ---------------------------------------------------------------------------
// Implicit-GEMM 3x3 conv + folded BN + ReLU.  K order = (tap, ci).
// Packed weight tiles (tkcnt KB, shared by all 8 waves) staged into LDS by
// the Tensor Data Mover; branchless clamped gathers for the image operand.
// ---------------------------------------------------------------------------
template <bool HAS2>
__global__ void conv3x3_wmma_kernel(const float* __restrict__ in, const float* __restrict__ in2,
                                    const _Float16* __restrict__ Wp,
                                    const float* __restrict__ bn_g, const float* __restrict__ bn_b,
                                    const float* __restrict__ bn_m, const float* __restrict__ bn_v,
                                    float* __restrict__ out, int Cinp, int Cout, int H, int W) {
  __shared__ _Float16 shA[36 * 512];  // up to 36 K-tiles (Cinp=128) * 1 KB
  const int cbcnt = Cinp >> 5;
  const int tkcnt = 9 * cbcnt;
  const _Float16* wsrc = Wp + (size_t)blockIdx.y * tkcnt * 512;

#ifdef HAVE_TDM
  if (threadIdx.x == 0) {
    // D# for a 1-D dword copy: tensor_dim0 = tile_dim0 = nd dwords (<= 9216).
    unsigned nd = (unsigned)(tkcnt * 256);
    unsigned lds = (unsigned)(uintptr_t)(void*)shA;           // LDS byte offset
    unsigned long long ga = (unsigned long long)(uintptr_t)wsrc;
    u32x4 g0;
    g0[0] = 1u;                                               // count=1, no gather
    g0[1] = lds;                                              // lds_addr [63:32]
    g0[2] = (unsigned)ga;                                     // global_addr [95:64]
    g0[3] = (unsigned)((ga >> 32) & 0x01FFFFFFu) | (2u << 30);// ga[56:32] | type=2
    i32x8 g1;
    g1[0] = (int)(2u << 16);                                  // data_size = 4 B
    g1[1] = (int)((nd & 0xFFFFu) << 16);                      // tensor_dim0 lo
    g1[2] = (int)(((nd >> 16) & 0xFFFFu) | (1u << 16));       // td0 hi | tensor_dim1=1
    g1[3] = (int)((nd & 0xFFFFu) << 16);                      // tile_dim0 = nd
    g1[4] = 0;
    g1[5] = (int)nd;                                          // tensor_dim0_stride
    g1[6] = 0;
    g1[7] = 0;
    i32x4 g2 = {0, 0, 0, 0};
    i32x4 g3 = {0, 0, 0, 0};
#if __clang_major__ >= 23
    i32x8 g4 = {0, 0, 0, 0, 0, 0, 0, 0};
    __builtin_amdgcn_tensor_load_to_lds(g0, g1, g2, g3, g4, 0);
#else
    __builtin_amdgcn_tensor_load_to_lds(g0, g1, g2, g3, 0);
#endif
    __builtin_amdgcn_s_wait_tensorcnt(0);
  }
#else
  {
    const float4* src = (const float4*)wsrc;
    float4* dst = (float4*)shA;
    int tot = tkcnt * 64;
    for (int i = threadIdx.x; i < tot; i += 256) dst[i] = src[i];
  }
#endif
  __syncthreads();

  const int lane = threadIdx.x & 31;
  const int wv = threadIdx.x >> 5;
  const int hi = lane >> 4, lo = lane & 15;
  const int b = blockIdx.z;
  const int m0 = blockIdx.y * 16;
  const int Npix = H * W;
  const int px = (blockIdx.x * 8 + wv) * 16 + lo;
  const int oh = px / W, ow = px % W;
  const float* inb  = in + (size_t)b * Cinp * Npix;
  const float* inb2 = HAS2 ? in2 + (size_t)b * Cinp * Npix : nullptr;

  v8f acc = {};
  #pragma unroll
  for (int t = 0; t < 9; ++t) {
    const int kh = t / 3, kw = t % 3;  // compile-time after unroll
    const int ih = oh + kh - 1, iw = ow + kw - 1;
    const bool pv = ((unsigned)ih < (unsigned)H) && ((unsigned)iw < (unsigned)W);
    const int ihc = ih < 0 ? 0 : (ih >= H ? H - 1 : ih);
    const int iwc = iw < 0 ? 0 : (iw >= W ? W - 1 : iw);
    const size_t offc = (size_t)ihc * W + iwc;
    for (int cb = 0; cb < cbcnt; ++cb) {
      const int tk = t * cbcnt + cb;
      v16h af = *(const v16h*)(shA + ((size_t)tk * 32 + lane) * 16);
      const int ci0 = (cb << 5) + hi * 16;
      const float* sp = inb + (size_t)ci0 * Npix + offc;
      __builtin_prefetch(sp + (size_t)32 * Npix, 0, 1);
      v16h bf;
      #pragma unroll
      for (int e = 0; e < 16; ++e) {
        float v = sp[(size_t)e * Npix];
        if (HAS2) v += inb2[(size_t)(ci0 + e) * Npix + offc];
        bf[e] = (_Float16)(pv ? v : 0.0f);
      }
      acc = __builtin_amdgcn_wmma_f32_16x16x32_f16(false, af, false, bf, (short)0, acc,
                                                   false, false);
    }
  }
  #pragma unroll
  for (int r = 0; r < 8; ++r) {
    int m = m0 + r + hi * 8;
    float sc = bn_g[m] * rsqrtf(bn_v[m] + BN_EPS);
    float sh = bn_b[m] - bn_m[m] * sc;
    float v = acc[r] * sc + sh;
    out[((size_t)b * Cout + m) * Npix + px] = v > 0.0f ? v : 0.0f;
  }
}

// Per (b, c) spatial mean of X[b, c, 0..1023].
__global__ void mean_kernel(const float* __restrict__ X, float* __restrict__ mean) {
  int idx = blockIdx.x * blockDim.x + threadIdx.x;
  if (idx >= 1024) return;
  const float4* p = (const float4*)(X + (size_t)idx * 1024);
  float s = 0.0f;
  for (int i = 0; i < 256; ++i) { float4 v = p[i]; s += v.x + v.y + v.z + v.w; }
  mean[idx] = s * (1.0f / 1024.0f);
}

// ---------------------------------------------------------------------------
// Covariance: cov[b] = X Xt / 1024 - mean meant.  M=N=128, K=1024, f16 WMMA.
// ---------------------------------------------------------------------------
__global__ void cov_wmma_kernel(const float* __restrict__ X, const float* __restrict__ mean,
                                float* __restrict__ cov) {
  const int lane = threadIdx.x & 31;
  const int hi = lane >> 4, lo = lane & 15;
  const int b = blockIdx.z;
  const int m0 = blockIdx.y * 16, n0 = blockIdx.x * 16;
  const float* arow = X + (size_t)b * 131072 + (size_t)(m0 + lo) * 1024;
  const float* brow = X + (size_t)b * 131072 + (size_t)(n0 + lo) * 1024;
  v8f acc = {};
  #pragma unroll 4
  for (int tk = 0; tk < 32; ++tk) {
    v16h af = load_a_frag_f32(arow + tk * 32, hi);
    v16h bf = load_b_frag_f32(brow + tk * 32, hi);
    acc = __builtin_amdgcn_wmma_f32_16x16x32_f16(false, af, false, bf, (short)0, acc,
                                                 false, false);
  }
  #pragma unroll
  for (int r = 0; r < 8; ++r) {
    int m = m0 + r + hi * 8, n = n0 + lo;
    cov[(size_t)b * 16384 + m * 128 + n] =
        acc[r] * (1.0f / 1024.0f) - mean[b * 128 + m] * mean[b * 128 + n];
  }
}

// trace + sqrt(trace) per batch.
__global__ void trace_kernel(const float* __restrict__ cov, float* __restrict__ tr,
                             float* __restrict__ sq) {
  int b = threadIdx.x;
  if (b >= 8) return;
  float s = 0.0f;
  for (int d = 0; d < 128; ++d) s += cov[(size_t)b * 16384 + d * 129];
  tr[b] = s;
  sq[b] = sqrtf(s);
}

// A1 = cov / tr ; Z0 = 0.5*(3I - A1)
__global__ void ns_prep_kernel(const float* __restrict__ cov, const float* __restrict__ tr,
                               float* __restrict__ A1, float* __restrict__ Z0) {
  int idx = blockIdx.x * blockDim.x + threadIdx.x;
  if (idx >= 8 * 16384) return;
  int b = idx >> 14;
  int ij = idx & 16383;
  int i = ij >> 7, j = ij & 127;
  float a = cov[idx] / tr[b];
  A1[idx] = a;
  Z0[idx] = 0.5f * ((i == j ? 3.0f : 0.0f) - a);
}

// ---------------------------------------------------------------------------
// Batched 128x128x128 Newton-Schulz matmul on V_WMMA_F32_16X16X4_F32 (fp32).
// All iterates are polynomials in the symmetric A1, hence symmetric:
// B[k][n] == B[n][k] -> both operands gathered as contiguous rows.
//   mode 0: D = A*B    mode 1: D = 0.5*(3I - A*B)    mode 2: D = (A*B)*sc[b]
// ---------------------------------------------------------------------------
__global__ void mm128_wmma_kernel(const float* __restrict__ A, const float* __restrict__ B,
                                  float* __restrict__ D, int mode, const float* __restrict__ sc) {
  const int lane = threadIdx.x & 31;
  const int hi = lane >> 4, lo = lane & 15;
  const int b = blockIdx.z;
  const int m0 = blockIdx.y * 16, n0 = blockIdx.x * 16;
  const size_t base = (size_t)b * 16384;
  const float* arow = A + base + (size_t)(m0 + lo) * 128;
  const float* brow = B + base + (size_t)(n0 + lo) * 128;  // symmetric B
  v8f acc = {};
#ifdef HAVE_WMMA_F32X4
  // A 16x4: lane(lo)=M row, VGPR j = K (hi*2 + j).  B 4x16: same via symmetry.
  #pragma unroll 8
  for (int k0 = 0; k0 < 128; k0 += 4) {
    v2f a = *(const v2f*)(arow + k0 + hi * 2);
    v2f bb = *(const v2f*)(brow + k0 + hi * 2);
    acc = __builtin_amdgcn_wmma_f32_16x16x4_f32(false, a, false, bb, (short)0, acc,
                                                false, false);
  }
#else
  #pragma unroll
  for (int tk = 0; tk < 4; ++tk) {
    v16h af = load_a_frag_f32(arow + tk * 32, hi);
    v16h bf = load_b_frag_f32(brow + tk * 32, hi);
    acc = __builtin_amdgcn_wmma_f32_16x16x32_f16(false, af, false, bf, (short)0, acc,
                                                 false, false);
  }
#endif
  #pragma unroll
  for (int r = 0; r < 8; ++r) {
    int m = m0 + r + hi * 8, n = n0 + lo;
    float v = acc[r];
    if (mode == 1) v = 0.5f * ((m == n ? 3.0f : 0.0f) - v);
    else if (mode == 2) v = v * sc[b];
    D[base + m * 128 + n] = v;
  }
}

// Upper-triangular extraction, row-major triu order.
__global__ void triu_kernel(const float* __restrict__ S, float* __restrict__ out) {
  int r = blockIdx.x, b = blockIdx.y, c = threadIdx.x;
  if (c < r) return;
  int t = r * 128 - (r * (r - 1)) / 2 + (c - r);
  out[(size_t)b * 8256 + t] = S[(size_t)b * 16384 + r * 128 + c];
}

// ---------------------------------------------------------------------------
// fc10: out[b,o] = relu(bn(s . W10[o,:] + bias[o])).  M=256, N=16(pad), K=8256.
// ---------------------------------------------------------------------------
__global__ void fc10_wmma_kernel(const _Float16* __restrict__ Wp, const float* __restrict__ s,
                                 const float* __restrict__ bias,
                                 const float* __restrict__ bn_g, const float* __restrict__ bn_b,
                                 const float* __restrict__ bn_m, const float* __restrict__ bn_v,
                                 float* __restrict__ out) {
  const int lane = threadIdx.x & 31;
  const int hi = lane >> 4, lo = lane & 15;
  const int m0 = blockIdx.x * 16;
  const int tkcnt = 8256 / 32;  // 258
  const float* srow = s + (size_t)(lo < 8 ? lo : 0) * 8256;
  const bool bvalid = lo < 8;
  v8f acc = {};
  #pragma unroll 2
  for (int tk = 0; tk < tkcnt; ++tk) {
    v16h af = *(const v16h*)(Wp + (((size_t)blockIdx.x * tkcnt + tk) * 32 + lane) * 16);
    v16h bf;
    if (bvalid) {
      bf = load_b_frag_f32(srow + tk * 32, hi);
    } else {
      #pragma unroll
      for (int e = 0; e < 16; ++e) bf[e] = (_Float16)0.0f;
    }
    acc = __builtin_amdgcn_wmma_f32_16x16x32_f16(false, af, false, bf, (short)0, acc,
                                                 false, false);
  }
  if (bvalid) {
    #pragma unroll
    for (int r = 0; r < 8; ++r) {
      int m = m0 + r + hi * 8;
      float sc = bn_g[m] * rsqrtf(bn_v[m] + BN_EPS);
      float sh = bn_b[m] - bn_m[m] * sc;
      float v = (acc[r] + bias[m]) * sc + sh;
      out[(size_t)lo * 256 + m] = v > 0.0f ? v : 0.0f;
    }
  }
}

// fc21: tiny 256->2 head.
__global__ void fc21_kernel(const float* __restrict__ sout, const float* __restrict__ w,
                            const float* __restrict__ bias, float* __restrict__ out) {
  int idx = threadIdx.x;
  if (idx >= 16) return;
  int b = idx >> 1, j = idx & 1;
  float s = bias[j];
  for (int i = 0; i < 256; ++i) s += sout[(size_t)b * 256 + i] * w[j * 256 + i];
  out[b * 2 + j] = s;
}

// ===========================================================================
extern "C" void kernel_launch(void* const* d_in, const int* in_sizes, int n_in,
                              void* d_out, int out_size, void* d_ws, size_t ws_size,
                              hipStream_t stream) {
  (void)in_sizes; (void)n_in; (void)out_size; (void)ws_size;
  auto F = [&](int i) { return (const float*)d_in[i]; };
  float* OUT = (float*)d_out;
  float* WSF = (float*)d_ws;

  // ---- workspace layout (float units) ----
  size_t cur = 0;
  auto falloc = [&](size_t n) { size_t o = cur; cur += (n + 63) & ~(size_t)63; return o; };
  const size_t o_x0 = falloc(16777216);  // HPF out, zero-padded to 32 channels
  const size_t o_x1 = falloc(16777216), o_x2 = falloc(16777216), o_x3 = falloc(16777216);
  const size_t o_x4 = falloc(16777216), o_x5 = falloc(16777216), o_x6 = falloc(16777216);
  const size_t o_P1 = falloc(4194304), o_T1 = falloc(8388608), o_T2 = falloc(8388608);
  const size_t o_P2 = falloc(2097152), o_T3 = falloc(2097152), o_T4 = falloc(2097152);
  const size_t o_P3 = falloc(524288),  o_T5 = falloc(1048576), o_SC = falloc(1048576);
  const size_t o_MEAN = falloc(1024), o_COV = falloc(131072), o_TR = falloc(64), o_SQ = falloc(64);
  const size_t o_MA1 = falloc(131072), o_MY = falloc(131072), o_MZ = falloc(131072);
  const size_t o_MT = falloc(131072), o_MY2 = falloc(131072), o_MZ2 = falloc(131072);
  const size_t o_TRIU = falloc(66048);
  const size_t o_PW = falloc(1609792);  // 3,219,456 halves
  _Float16* PW = (_Float16*)(WSF + o_PW);

  const size_t hw_cb[6] = {0, 9216, 18432, 27648, 36864, 46080};
  const size_t hw_sfe_base = 55296, hw_sfe_stride = 350208;
  const size_t hw_in_sfe[6] = {0, 18432, 55296, 92160, 129024, 202752};
  const size_t hw_fc10 = hw_sfe_base + 3 * hw_sfe_stride;

  auto repackc = [&](const float* Wsrc, size_t hoff, int M, int Cin) {
    int Cinp = (Cin + 31) & ~31;
    int total = M * 9 * Cinp;
    repack_conv_kernel<<<(total + 255) / 256, 256, 0, stream>>>(Wsrc, PW + hoff, M, Cin, Cinp);
  };
  auto convL = [&](const float* in, const float* in2, size_t hoff, int pbase, float* out,
                   int Cinp, int Cout, int H, int W) {
    dim3 g(H * W / 128, Cout / 16, 8);
    if (in2)
      conv3x3_wmma_kernel<true><<<g, 256, 0, stream>>>(in, in2, PW + hoff,
          F(pbase + 1), F(pbase + 2), F(pbase + 3), F(pbase + 4), out, Cinp, Cout, H, W);
    else
      conv3x3_wmma_kernel<false><<<g, 256, 0, stream>>>(in, nullptr, PW + hoff,
          F(pbase + 1), F(pbase + 2), F(pbase + 3), F(pbase + 4), out, Cinp, Cout, H, W);
  };
  auto pool = [&](const float* in, float* out, int C, int Hin) {
    size_t total = (size_t)8 * C * (Hin / 2) * (Hin / 2);
    avgpool_kernel<<<(unsigned)((total + 255) / 256), 256, 0, stream>>>(in, out, 8, C, Hin, Hin);
  };
  auto mm = [&](const float* A, const float* B, float* D, int mode, const float* sc) {
    mm128_wmma_kernel<<<dim3(8, 8, 8), 32, 0, stream>>>(A, B, D, mode, sc);
  };

  // ---- 1. repack all GEMM weights ----
  for (int i = 0; i < 6; ++i) repackc(F(2 + 5 * i), hw_cb[i], 32, (i == 0 ? 30 : 32));
  for (int s = 0; s < 3; ++s) {
    int sb = 32 + s * 30;
    size_t hb = hw_sfe_base + s * hw_sfe_stride;
    repackc(F(sb + 0),  hb + hw_in_sfe[0], 64, 32);
    repackc(F(sb + 5),  hb + hw_in_sfe[1], 64, 64);
    repackc(F(sb + 10), hb + hw_in_sfe[2], 64, 64);
    repackc(F(sb + 15), hb + hw_in_sfe[3], 64, 64);
    repackc(F(sb + 20), hb + hw_in_sfe[4], 128, 64);
    repackc(F(sb + 25), hb + hw_in_sfe[5], 128, 128);
  }
  repack_flat_kernel<<<(256 * 8256 + 255) / 256, 256, 0, stream>>>(F(122), PW + hw_fc10, 256, 8256);

  // ---- 2. HPF + TLU (32-channel zero-padded output) ----
  hpf_kernel<<<65536, 256, 0, stream>>>(F(0), F(1), WSF + o_x0, 8, 256, 256);

  // ---- 3. backbone convs ----
  convL(WSF + o_x0, nullptr, hw_cb[0], 2,  WSF + o_x1, 32, 32, 256, 256);
  convL(WSF + o_x1, nullptr, hw_cb[1], 7,  WSF + o_x2, 32, 32, 256, 256);
  convL(WSF + o_x2, nullptr, hw_cb[2], 12, WSF + o_x3, 32, 32, 256, 256);
  convL(WSF + o_x3, nullptr, hw_cb[3], 17, WSF + o_x4, 32, 32, 256, 256);

  auto sfe = [&](const float* xin, int s) {
    int sb = 32 + s * 30;
    size_t hb = hw_sfe_base + s * hw_sfe_stride;
    pool(xin, WSF + o_P1, 32, 256);
    convL(WSF + o_P1, nullptr, hb + hw_in_sfe[0], sb + 0,  WSF + o_T1, 32, 64, 128, 128);
    convL(WSF + o_T1, nullptr, hb + hw_in_sfe[1], sb + 5,  WSF + o_T2, 64, 64, 128, 128);
    pool(WSF + o_T2, WSF + o_P2, 64, 128);
    convL(WSF + o_P2, nullptr, hb + hw_in_sfe[2], sb + 10, WSF + o_T3, 64, 64, 64, 64);
    convL(WSF + o_T3, nullptr, hb + hw_in_sfe[3], sb + 15, WSF + o_T4, 64, 64, 64, 64);
    pool(WSF + o_T4, WSF + o_P3, 64, 64);
    convL(WSF + o_P3, nullptr, hb + hw_in_sfe[4], sb + 20, WSF + o_T5, 64, 128, 32, 32);
    convL(WSF + o_T5, nullptr, hb + hw_in_sfe[5], sb + 25, WSF + o_SC, 128, 128, 32, 32);
  };

  auto branch = [&](int bi) {
    const float* X = WSF + o_SC;
    mean_kernel<<<4, 256, 0, stream>>>(X, WSF + o_MEAN);
    cov_wmma_kernel<<<dim3(8, 8, 8), 32, 0, stream>>>(X, WSF + o_MEAN, WSF + o_COV);
    trace_kernel<<<1, 8, 0, stream>>>(WSF + o_COV, WSF + o_TR, WSF + o_SQ);
    ns_prep_kernel<<<512, 256, 0, stream>>>(WSF + o_COV, WSF + o_TR, WSF + o_MA1, WSF + o_MZ);
    float *A1 = WSF + o_MA1, *Y = WSF + o_MY, *Z = WSF + o_MZ;
    float *T = WSF + o_MT, *Y2 = WSF + o_MY2, *Z2 = WSF + o_MZ2;
    mm(A1, Z, Y, 0, nullptr);                       // Y0 = A1 * Z0
    for (int it = 0; it < 3; ++it) {                // iterN - 2 = 3
      mm(Z, Y, T, 1, nullptr);                      // T = 0.5*(3I - Z*Y)
      mm(Y, T, Y2, 0, nullptr);                     // Y' = Y*T
      mm(T, Z, Z2, 0, nullptr);                     // Z' = T*Z
      float* t1 = Y; Y = Y2; Y2 = t1;
      float* t2 = Z; Z = Z2; Z2 = t2;
    }
    mm(Z, Y, T, 1, nullptr);                        // T = 0.5*(3I - Z*Y)
    mm(Y, T, A1, 2, WSF + o_SQ);                    // out = Y*T * sqrt(trace)
    triu_kernel<<<dim3(128, 8), 128, 0, stream>>>(A1, WSF + o_TRIU);
    float* sout = OUT + 48 + (size_t)bi * 2048;
    fc10_wmma_kernel<<<16, 32, 0, stream>>>(PW + hw_fc10, WSF + o_TRIU, F(123),
                                            F(124), F(125), F(126), F(127), sout);
    fc21_kernel<<<1, 16, 0, stream>>>(sout, F(128), F(129), OUT + (size_t)bi * 16);
  };

  // ---- scale 1 ----
  sfe(WSF + o_x4, 0);
  branch(0);
  // ---- x5 = cbr(x3 + x4), scale 2 ----
  convL(WSF + o_x3, WSF + o_x4, hw_cb[4], 22, WSF + o_x5, 32, 32, 256, 256);
  sfe(WSF + o_x5, 1);
  branch(1);
  // ---- x6 = cbr(x2 + x5), scale 3 ----
  convL(WSF + o_x2, WSF + o_x5, hw_cb[5], 27, WSF + o_x6, 32, 32, 256, 256);
  sfe(WSF + o_x6, 2);
  branch(2);
}
